// LocalSpatioTemporalAttentionPatch_58119497450439
// MI455X (gfx1250) — matrix operand order repaired and checked by reading, not compiled, not run on hardware
//
#include <hip/hip_runtime.h>
#include <hip/hip_bf16.h>

typedef __attribute__((ext_vector_type(16))) _Float16 v16h;
typedef __attribute__((ext_vector_type(8)))  _Float16 v8h;
typedef __attribute__((ext_vector_type(4)))  _Float16 v4h;
typedef __attribute__((ext_vector_type(8)))  float    v8f;

#define T_DIM 3
#define HH    96
#define WW    96
#define C_DIM 128
#define NPOS  (HH * WW)        // 9216
#define NROWS (T_DIM * NPOS)   // 27648
#define NHEADS 8
#define DHEAD  16
#define P_DIM  25
#define NKEYS  (T_DIM * P_DIM) // 75

__device__ __forceinline__ int iclamp(int v, int lo, int hi) {
    return v < lo ? lo : (v > hi ? hi : v);
}

// ---------------------------------------------------------------------------
// Prep 1: convert x to f16 (L2-resident activation copy, halves GEMM traffic).
// ---------------------------------------------------------------------------
__global__ __launch_bounds__(256) void cvt_x_kernel(
    const float* __restrict__ x, _Float16* __restrict__ Xh)
{
    const int i = (blockIdx.x * blockDim.x + threadIdx.x) * 4;
    const float4 v = *(const float4*)(x + i);
    v4h h;
    h[0] = (_Float16)v.x; h[1] = (_Float16)v.y;
    h[2] = (_Float16)v.z; h[3] = (_Float16)v.w;
    *(v4h*)(Xh + i) = h;
}

// ---------------------------------------------------------------------------
// Prep 2: convert + transpose weights to f16 column-major:
//   Wt [384][128]: Wt[cg*128 + k] = W{cg/128}[k][cg%128]   (cg = QKV channel)
//   Wot[128][128]: Wot[n*128 + k] = Wo[k][n]
// Makes every WMMA B-fragment two contiguous 16B loads.
// ---------------------------------------------------------------------------
__global__ __launch_bounds__(256) void prep_w_kernel(
    const float* __restrict__ Wq, const float* __restrict__ Wk,
    const float* __restrict__ Wv, const float* __restrict__ Wo,
    _Float16* __restrict__ Wt, _Float16* __restrict__ Wot)
{
    const int i = blockIdx.x * blockDim.x + threadIdx.x;   // 0..65535
    if (i < 3 * C_DIM * C_DIM) {                            // 49152: QKV
        const int cg  = i >> 7;        // 0..383
        const int k   = i & 127;
        const int mat = cg >> 7;       // 0:Q 1:K 2:V
        const int n   = cg & 127;
        const float* W = (mat == 0) ? Wq : ((mat == 1) ? Wk : Wv);
        Wt[i] = (_Float16)W[k * C_DIM + n];
    } else {
        const int j = i - 3 * C_DIM * C_DIM;               // 0..16383
        const int n = j >> 7, k = j & 127;
        Wot[j] = (_Float16)Wo[k * C_DIM + n];
    }
}

// ---------------------------------------------------------------------------
// Shared GEMM helpers (WMMA fragment builders).
// A (16x32 f16): lane L holds row M=L&15; elements = runs [k1..k1+7],[k1+16..k1+23],
//   k1 = kb + 8*(L>=16).  B (32x16): lane holds col N=L&15, same k runs as A.
// ---------------------------------------------------------------------------
__device__ __forceinline__ v16h frag_pair(v8h lo, v8h hi) {
    v16h a;
#pragma unroll
    for (int i = 0; i < 8; ++i) { a[i] = lo[i]; a[8 + i] = hi[i]; }
    return a;
}

// ---------------------------------------------------------------------------
// Kernel A: fused QKV projection. One block per 16-row tile; the 4KB f16
// A-tile is staged into LDS via global_load_async_to_lds_b128 and reused by
// all 8 waves x 3 col-tiles (24 tiles = [Wq|Wk|Wv]). Embedding adds fused.
// ---------------------------------------------------------------------------
__global__ __launch_bounds__(256) void qkv_proj_kernel(
    const _Float16* __restrict__ Xh, const _Float16* __restrict__ Wt,
    const float* __restrict__ temp_emb, const float* __restrict__ sp_emb,
    _Float16* __restrict__ Qw, _Float16* __restrict__ Kw,
    _Float16* __restrict__ Vw)
{
    __shared__ __align__(16) _Float16 tA[16 * C_DIM];      // 4KB row tile

    const int tid  = threadIdx.x;
    const int wv   = tid >> 5;          // wave 0..7
    const int lane = tid & 31;
    const int row0 = blockIdx.x * 16;

    // Async-stage the A tile: 256 threads x 16B = 4KB, no VGPR round-trip.
    {
        const unsigned ldsdst =
            (unsigned)(uintptr_t)(&tA[0]) + (unsigned)tid * 16u;
        const void* gsrc = (const void*)(Xh + (size_t)row0 * C_DIM + tid * 8);
        asm volatile("global_load_async_to_lds_b128 %0, %1, off"
                     :: "v"(ldsdst), "v"(gsrc) : "memory");
        asm volatile("s_wait_asynccnt 0x0" ::: "memory");
    }
    __syncthreads();

    const int half = lane >> 4;
    const int lm   = lane & 15;
    const _Float16* arow = &tA[lm * C_DIM];

#pragma unroll
    for (int c3 = 0; c3 < 3; ++c3) {
        const int ctg = wv + c3 * 8;            // global col tile 0..23
        const int mat = ctg >> 3;               // 0:Q 1:K 2:V
        const int ncl = (ctg & 7) * 16 + lm;    // channel within matrix
        const _Float16* brow = Wt + (size_t)(ctg * 16 + lm) * C_DIM;

        v8f acc = {};
#pragma unroll
        for (int ch = 0; ch < 4; ++ch) {        // K = 128 = 4 x 32
            const int kb = ch * 32;
            const int k1 = kb + half * 8;
            const int k2 = kb + half * 8;
            const v16h a = frag_pair(*(const v8h*)(arow + k1),
                                     *(const v8h*)(arow + k1 + 16));
            const v16h b = frag_pair(*(const v8h*)(brow + k2),
                                     *(const v8h*)(brow + k2 + 16));
            acc = __builtin_amdgcn_wmma_f32_16x16x32_f16(
                false, a, false, b, (short)0, acc, false, false);
        }

        // D VGPR r -> (M = r + 8*half, N = lane&15); fuse embedding adds.
#pragma unroll
        for (int r = 0; r < 8; ++r) {
            const int row = row0 + r + half * 8;
            const int t   = row / NPOS;
            const int pos = row - t * NPOS;
            const int yq  = pos / WW;
            const int xq  = pos - yq * WW;
            float v = acc[r];
            if (mat == 0) {
                const int yr = yq - iclamp(yq, 2, HH - 3) + 2;
                const int xr = xq - iclamp(xq, 2, WW - 3) + 2;
                const int qi = yr * 5 + xr;
                v += temp_emb[t * C_DIM + ncl] + sp_emb[qi * C_DIM + ncl];
                Qw[(size_t)row * C_DIM + ncl] = (_Float16)v;
            } else if (mat == 1) {
                v += temp_emb[t * C_DIM + ncl];
                Kw[(size_t)row * C_DIM + ncl] = (_Float16)v;
            } else {
                Vw[(size_t)row * C_DIM + ncl] = (_Float16)v;
            }
        }
    }
}

// ---------------------------------------------------------------------------
// Kernel B: local attention. One wave per (position, head); lanes own keys
// (75 keys in 3 chunks); fp32 math, wave-reduce softmax via shfl_xor.
// ---------------------------------------------------------------------------
__global__ __launch_bounds__(256) void attn_kernel(
    const _Float16* __restrict__ Qw, const _Float16* __restrict__ Kw,
    const _Float16* __restrict__ Vw, const float* __restrict__ sp_emb,
    _Float16* __restrict__ Ow)
{
    const int wid  = (blockIdx.x * blockDim.x + threadIdx.x) >> 5;
    const int lane = threadIdx.x & 31;
    const int head = wid & 7;
    const int pos  = wid >> 3;          // 0..9215
    const int yq = pos / WW, xq = pos - (pos / WW) * WW;
    const int yc = iclamp(yq, 2, HH - 3);
    const int xc = iclamp(xq, 2, WW - 3);
    const int cb = head * DHEAD;

    // Broadcast-load query vectors (b128 pairs, same addr across lanes).
    float q[T_DIM][DHEAD];
#pragma unroll
    for (int t = 0; t < T_DIM; ++t) {
        const _Float16* qp = Qw + ((size_t)(t * NPOS + pos)) * C_DIM + cb;
        const v8h q0 = *(const v8h*)qp;
        const v8h q1 = *(const v8h*)(qp + 8);
#pragma unroll
        for (int d = 0; d < 8; ++d) {
            q[t][d] = (float)q0[d];
            q[t][8 + d] = (float)q1[d];
        }
    }

    float s[3][T_DIM];
#pragma unroll
    for (int c = 0; c < 3; ++c) {
        const int n = c * 32 + lane;
        const bool valid = (n < NKEYS);
        const int nn = valid ? n : 0;
        const int tk = nn / P_DIM;
        const int p  = nn - tk * P_DIM;
        const int pr = p / 5, pc = p - pr * 5;
        const int yy = yc + pr - 2, xx = xc + pc - 2;
        const _Float16* kp =
            Kw + ((size_t)(tk * NPOS + yy * WW + xx)) * C_DIM + cb;
        const v8h k0 = *(const v8h*)kp;
        const v8h k1 = *(const v8h*)(kp + 8);
        const float* se = sp_emb + p * C_DIM + cb;
        float a0 = 0.f, a1 = 0.f, a2 = 0.f;
#pragma unroll
        for (int d = 0; d < DHEAD; ++d) {
            const float kv =
                (float)(d < 8 ? k0[d & 7] : k1[d & 7]) + se[d];
            a0 += q[0][d] * kv;
            a1 += q[1][d] * kv;
            a2 += q[2][d] * kv;
        }
        s[c][0] = valid ? a0 * 0.25f : -1.0e30f;
        s[c][1] = valid ? a1 * 0.25f : -1.0e30f;
        s[c][2] = valid ? a2 * 0.25f : -1.0e30f;
    }

#pragma unroll
    for (int t = 0; t < T_DIM; ++t) {
        float m = fmaxf(s[0][t], fmaxf(s[1][t], s[2][t]));
#pragma unroll
        for (int off = 16; off > 0; off >>= 1)
            m = fmaxf(m, __shfl_xor(m, off, 32));
        float sum = 0.f;
#pragma unroll
        for (int c = 0; c < 3; ++c) {
            s[c][t] = __expf(s[c][t] - m);   // invalid lanes underflow to 0
            sum += s[c][t];
        }
#pragma unroll
        for (int off = 16; off > 0; off >>= 1)
            sum += __shfl_xor(sum, off, 32);
        const float inv = 1.0f / sum;
#pragma unroll
        for (int c = 0; c < 3; ++c) s[c][t] *= inv;
    }

    float o[T_DIM][DHEAD] = {};
#pragma unroll
    for (int c = 0; c < 3; ++c) {
        const int n  = c * 32 + lane;
        const int nn = (n < NKEYS) ? n : 0;   // weight already 0 if invalid
        const int tk = nn / P_DIM;
        const int p  = nn - tk * P_DIM;
        const int pr = p / 5, pc = p - pr * 5;
        const int yy = yc + pr - 2, xx = xc + pc - 2;
        const _Float16* vp =
            Vw + ((size_t)(tk * NPOS + yy * WW + xx)) * C_DIM + cb;
        const v8h v0 = *(const v8h*)vp;
        const v8h v1 = *(const v8h*)(vp + 8);
#pragma unroll
        for (int d = 0; d < DHEAD; ++d) {
            const float vv = (float)(d < 8 ? v0[d & 7] : v1[d & 7]);
            o[0][d] += s[c][0] * vv;
            o[1][d] += s[c][1] * vv;
            o[2][d] += s[c][2] * vv;
        }
    }
#pragma unroll
    for (int t = 0; t < T_DIM; ++t)
#pragma unroll
        for (int d = 0; d < DHEAD; ++d)
#pragma unroll
            for (int off = 16; off > 0; off >>= 1)
                o[t][d] += __shfl_xor(o[t][d], off, 32);

    if (lane == 0) {
#pragma unroll
        for (int t = 0; t < T_DIM; ++t)
#pragma unroll
            for (int d = 0; d < DHEAD; ++d)
                Ow[((size_t)(t * NPOS + pos)) * C_DIM + cb + d] =
                    (_Float16)o[t][d];
    }
}

// ---------------------------------------------------------------------------
// Kernel C: output projection + fused transpose to (B,H,W,T*C).
// One block per 16-row tile (async LDS stage), 8 waves = 8 col tiles.
// ---------------------------------------------------------------------------
__global__ __launch_bounds__(256) void out_proj_kernel(
    const _Float16* __restrict__ Ow, const _Float16* __restrict__ Wot,
    float* __restrict__ out)
{
    __shared__ __align__(16) _Float16 tA[16 * C_DIM];

    const int tid  = threadIdx.x;
    const int wv   = tid >> 5;
    const int lane = tid & 31;
    const int row0 = blockIdx.x * 16;

    {
        const unsigned ldsdst =
            (unsigned)(uintptr_t)(&tA[0]) + (unsigned)tid * 16u;
        const void* gsrc = (const void*)(Ow + (size_t)row0 * C_DIM + tid * 8);
        asm volatile("global_load_async_to_lds_b128 %0, %1, off"
                     :: "v"(ldsdst), "v"(gsrc) : "memory");
        asm volatile("s_wait_asynccnt 0x0" ::: "memory");
    }
    __syncthreads();

    const int half = lane >> 4;
    const int lm   = lane & 15;
    const int ncl  = wv * 16 + lm;
    const _Float16* arow = &tA[lm * C_DIM];
    const _Float16* brow = Wot + (size_t)ncl * C_DIM;

    v8f acc = {};
#pragma unroll
    for (int ch = 0; ch < 4; ++ch) {
        const int kb = ch * 32;
        const int k1 = kb + half * 8;
        const int k2 = kb + half * 8;
        const v16h a = frag_pair(*(const v8h*)(arow + k1),
                                 *(const v8h*)(arow + k1 + 16));
        const v16h b = frag_pair(*(const v8h*)(brow + k2),
                                 *(const v8h*)(brow + k2 + 16));
        acc = __builtin_amdgcn_wmma_f32_16x16x32_f16(
            false, a, false, b, (short)0, acc, false, false);
    }

#pragma unroll
    for (int r = 0; r < 8; ++r) {
        const int row = row0 + r + half * 8;
        const int t   = row / NPOS;
        const int pos = row - t * NPOS;  // y*96 + x
        out[((size_t)pos * T_DIM + t) * C_DIM + ncl] = acc[r];
    }
}

// ---------------------------------------------------------------------------
extern "C" void kernel_launch(void* const* d_in, const int* in_sizes, int n_in,
                              void* d_out, int out_size, void* d_ws,
                              size_t ws_size, hipStream_t stream)
{
    (void)in_sizes; (void)n_in; (void)out_size; (void)ws_size;
    const float* x        = (const float*)d_in[0];
    const float* Wq       = (const float*)d_in[1];
    const float* Wk       = (const float*)d_in[2];
    const float* Wv       = (const float*)d_in[3];
    const float* Wo       = (const float*)d_in[4];
    const float* temp_emb = (const float*)d_in[5];
    const float* sp_emb   = (const float*)d_in[6];
    float* out = (float*)d_out;

    const size_t n = (size_t)NROWS * C_DIM;   // 3,538,944
    _Float16* Qw  = (_Float16*)d_ws;
    _Float16* Kw  = Qw + n;
    _Float16* Vw  = Kw + n;
    _Float16* Ow  = Vw + n;
    _Float16* Xh  = Ow + n;
    _Float16* Wt  = Xh + n;                    // 384*128
    _Float16* Wot = Wt + 3 * C_DIM * C_DIM;    // 128*128  (~35.5 MB total)

    cvt_x_kernel<<<3456, 256, 0, stream>>>(x, Xh);          // 3538944/4 thr
    prep_w_kernel<<<256, 256, 0, stream>>>(Wq, Wk, Wv, Wo, Wt, Wot);
    qkv_proj_kernel<<<1728, 256, 0, stream>>>(Xh, Wt, temp_emb, sp_emb,
                                              Qw, Kw, Vw);
    attn_kernel<<<9216, 256, 0, stream>>>(Qw, Kw, Vw, sp_emb, Ow);
    out_proj_kernel<<<1728, 256, 0, stream>>>(Ow, Wot, out);
}
